// BotNetwork_14688788152878
// MI455X (gfx1250) — matrix operand order repaired and checked by reading, not compiled, not run on hardware
//
#include <hip/hip_runtime.h>
#include <hip/hip_bf16.h>

typedef __attribute__((ext_vector_type(16))) _Float16 v16h;
typedef __attribute__((ext_vector_type(8)))  _Float16 v8h;
typedef __attribute__((ext_vector_type(8)))  float    v8f;
typedef __attribute__((ext_vector_type(4)))  float    v4f;

#define WAVES 2            // waves per workgroup
#define MROWS 32           // batch rows per wave (two 16-row WMMA tiles)
#define GRIDN 11

// packed f16 weight offsets inside d_ws (units: halfs)
#define OFF_BOT   0        // 128x64
#define OFF_CREW  8192     // 128x64
#define OFF_SHIP  16384    // 128x64
#define OFF_FC1   24576    // 192x128
#define OFF_FC2   49152    // 128x64
#define OFF_OUT   57344    // 64x16

// ---------------------------------------------------------------------------
// Pack fp32 weight [K x N] into WMMA B-fragment order, f16, K padded to Kpad,
// N padded to Npad. Fragment f covers (ktile,ntile); within a fragment, lane l
// holds column n = nt*16 + (l&15), halfs h=0..15 are k = kt*32 + 8*(l>>4) + (h&7) + 16*(h>>3).
// Each lane's 16 halfs are one contiguous 32-byte span -> single vector load.
// ---------------------------------------------------------------------------
__global__ void pack_weights_kernel(const float* __restrict__ w, _Float16* __restrict__ dst,
                                    int K, int N, int Kpad, int Npad) {
  int total  = Kpad * Npad;
  int ntiles = Npad >> 4;
  for (int i = blockIdx.x * blockDim.x + threadIdx.x; i < total; i += gridDim.x * blockDim.x) {
    int f    = i >> 9;            // 512 halfs per 32x16 fragment
    int r    = i & 511;
    int lane = r >> 4;
    int h    = r & 15;
    int kt = f / ntiles;
    int nt = f - kt * ntiles;
    int k  = kt * 32 + (h & 7) + ((lane >> 4) << 3) + ((h >> 3) << 4);
    int n  = nt * 16 + (lane & 15);
    float v = (k < K && n < N) ? w[k * N + n] : 0.0f;
    dst[i] = (_Float16)v;
  }
}

// ---------------------------------------------------------------------------
// Main fused MLP kernel: 1 wave == 32 batch rows (two 16-row WMMA M-tiles,
// so every B fragment load is reused by two v_wmma ops).
// ---------------------------------------------------------------------------
struct WaveBuf {
  _Float16 act[MROWS * 128];   // encoder input staging / fc1 output (32x128)
  _Float16 enc[MROWS * 192];   // concat encoder output / fc2 output (32x192)
  int      bpos[MROWS];
  int      cpos[MROWS];
  unsigned mask[MROWS];
};

// A fragment (16x32 f16) from row-major LDS tile of width W.
// ISA layout: lanes 0-15 -> M=lane, K = kt*32 + {0..7, 16..23};
//             lanes 16-31 -> M=lane-16, K = kt*32 + {8..15, 24..31}.
// Both K octets are contiguous -> two ds_load_b128.
__device__ inline v16h load_a_frag(const _Float16* in, int W, int ktile, int lane) {
  int m  = lane & 15;
  int hi = lane >> 4;
  const _Float16* base = in + m * W + ktile * 32 + hi * 8;
  union { v16h v; v8h h[2]; } u;
  u.h[0] = *(const v8h*)(base);
  u.h[1] = *(const v8h*)(base + 16);
  return u.v;
}

// B fragment: 32 contiguous bytes per lane from the packed weight buffer.
__device__ inline v16h load_b_frag(const _Float16* __restrict__ wf, int frag, int lane) {
  return *(const v16h*)(wf + ((size_t)((frag << 5) + lane) << 4));
}

// 32xK @ KxN GEMM (KT ktiles of 32, NT ntiles of 16) + bias + relu,
// result written as f16 into dst (row-major, width DW, column offset co).
// Two M-tiles share each B fragment.
template <int KT, int NT>
__device__ inline void gemm_relu_to_lds(const _Float16* __restrict__ in, int W,
                                        const _Float16* __restrict__ wf,
                                        const float* __restrict__ bias,
                                        _Float16* __restrict__ dst, int DW, int co,
                                        int lane) {
  const v8f vz = {0.f, 0.f, 0.f, 0.f, 0.f, 0.f, 0.f, 0.f};
  v8f acc0[NT], acc1[NT];
#pragma unroll
  for (int n = 0; n < NT; ++n) { acc0[n] = vz; acc1[n] = vz; }
#pragma unroll
  for (int k = 0; k < KT; ++k) {
    v16h a0 = load_a_frag(in, W, k, lane);
    v16h a1 = load_a_frag(in + 16 * W, W, k, lane);
#pragma unroll
    for (int n = 0; n < NT; ++n) {
      v16h b = load_b_frag(wf, k * NT + n, lane);
      acc0[n] = __builtin_amdgcn_wmma_f32_16x16x32_f16(false, a0, false, b,
                                                       (short)0, acc0[n], false, false);
      acc1[n] = __builtin_amdgcn_wmma_f32_16x16x32_f16(false, a1, false, b,
                                                       (short)0, acc1[n], false, false);
    }
  }
  int nl = lane & 15;
  int hi = lane >> 4;
#pragma unroll
  for (int n = 0; n < NT; ++n) {
    float bv = bias[n * 16 + nl];
#pragma unroll
    for (int v = 0; v < 8; ++v) {
      int m = v + 8 * hi;                 // C layout: VGPR v -> row v + 8*hi, col nl
      float x0 = acc0[n][v] + bv;
      float x1 = acc1[n][v] + bv;
      x0 = x0 > 0.f ? x0 : 0.f;
      x1 = x1 > 0.f ? x1 : 0.f;
      dst[m * DW + co + n * 16 + nl]        = (_Float16)x0;
      dst[(16 + m) * DW + co + n * 16 + nl] = (_Float16)x1;
    }
  }
}

// Stage one 32x121 fp32 tile (3872 contiguous, 16B-aligned floats) into a
// 32x128 f16 LDS tile (cols 121..127 zero). Non-temporal: inputs are
// streamed exactly once; keep the hot weight set cache-resident.
__device__ inline void stage_tile(const float* __restrict__ src, _Float16* __restrict__ act, int lane) {
  const v4f* s4 = (const v4f*)src;
  for (int i = lane; i < (MROWS * 121) / 4; i += 32) {
    v4f q = __builtin_nontemporal_load(&s4[i]);
    int e = i << 2;
#pragma unroll
    for (int j = 0; j < 4; ++j) {
      int idx = e + j;
      int r = idx / 121;
      int c = idx - r * 121;
      act[r * 128 + c] = (_Float16)q[j];
    }
  }
  for (int i = lane; i < MROWS * 7; i += 32) {
    int r = i / 7;
    int c = 121 + (i - r * 7);
    act[r * 128 + c] = (_Float16)0.f;
  }
}

// First-occurrence argmax over 121 fp32 per row; 2 lanes per row + shfl
// combine; processes 16 rows per call (rows ro..ro+15).
__device__ inline void tile_argmax(const float* __restrict__ src, int* __restrict__ pos,
                                   int ro, int lane) {
  int row  = ro + (lane >> 1);
  int part = lane & 1;
  const float* rp = src + (size_t)row * 121;
  int start = part ? 64 : 0;
  int cnt   = part ? 57 : 64;
  float m = -__builtin_inff();
  int   mi = start;
  for (int i = 0; i < cnt; ++i) {
    float v = rp[start + i];
    if (v > m) { m = v; mi = start + i; }
  }
  float om  = __shfl_xor(m, 1);
  int   omi = __shfl_xor(mi, 1);
  if (om > m || (om == m && omi < mi)) mi = omi;
  if (part == 0) pos[row] = mi;
}

// One lane per row (32 rows == 32 lanes).
__device__ inline void compute_mask(const _Float16* __restrict__ shipAct,
                                    const int* __restrict__ bpos, const int* __restrict__ cpos,
                                    unsigned* __restrict__ msk, int lane) {
  int bp = bpos[lane], cp = cpos[lane];
  int bx = bp / GRIDN, by = bp - bx * GRIDN;
  int cx = cp / GRIDN, cy = cp - cx * GRIDN;
  const int dx[9] = {0, 0, 0, 1, -1, 1, 1, -1, -1};
  const int dy[9] = {0, 1, -1, 0, 0, 1, -1, 1, -1};
  unsigned m = 0;
#pragma unroll
  for (int a = 0; a < 9; ++a) {
    int nx = bx + dx[a], ny = by + dy[a];
    bool inb = (nx >= 0) & (nx < GRIDN) & (ny >= 0) & (ny < GRIDN);
    int fx = nx < 0 ? 0 : (nx > GRIDN - 1 ? GRIDN - 1 : nx);
    int fy = ny < 0 ? 0 : (ny > GRIDN - 1 ? GRIDN - 1 : ny);
    float sv = (float)shipAct[lane * 128 + fx * GRIDN + fy];  // ship is 0/1, exact in f16
    bool ok = inb && (sv == 1.0f) && !((nx == cx) & (ny == cy));
    if (ok) m |= (1u << a);
  }
  msk[lane] = m;
}

__global__ __launch_bounds__(32 * WAVES) void botnet_kernel(
    const float* __restrict__ bot, const float* __restrict__ crew, const float* __restrict__ ship,
    const float* __restrict__ bot_b, const float* __restrict__ crew_b, const float* __restrict__ ship_b,
    const float* __restrict__ fc1_b, const float* __restrict__ fc2_b, const float* __restrict__ out_b,
    const _Float16* __restrict__ ws, float* __restrict__ out) {
  __shared__ WaveBuf sb[WAVES];
  int lane = threadIdx.x & 31;
  int wave = threadIdx.x >> 5;
  WaveBuf& w = sb[wave];
  size_t rowBase = ((size_t)blockIdx.x * WAVES + wave) * MROWS;

  // --- encoders ---  (per-wave private LDS; DS ops are in-order within a wave)
  stage_tile(bot + rowBase * 121, w.act, lane);
  tile_argmax(bot + rowBase * 121, w.bpos, 0, lane);
  tile_argmax(bot + rowBase * 121, w.bpos, 16, lane);
  gemm_relu_to_lds<4, 4>(w.act, 128, ws + OFF_BOT, bot_b, w.enc, 192, 0, lane);

  stage_tile(crew + rowBase * 121, w.act, lane);
  tile_argmax(crew + rowBase * 121, w.cpos, 0, lane);
  tile_argmax(crew + rowBase * 121, w.cpos, 16, lane);
  gemm_relu_to_lds<4, 4>(w.act, 128, ws + OFF_CREW, crew_b, w.enc, 192, 64, lane);

  stage_tile(ship + rowBase * 121, w.act, lane);
  compute_mask(w.act, w.bpos, w.cpos, w.mask, lane);
  gemm_relu_to_lds<4, 4>(w.act, 128, ws + OFF_SHIP, ship_b, w.enc, 192, 128, lane);

  // --- fc1: [32,192] @ [192,128] -> act ---
  gemm_relu_to_lds<6, 8>(w.enc, 192, ws + OFF_FC1, fc1_b, w.act, 128, 0, lane);
  // --- fc2: [32,128] @ [128,64] -> enc[:, 0:64] ---
  gemm_relu_to_lds<4, 4>(w.act, 128, ws + OFF_FC2, fc2_b, w.enc, 192, 0, lane);

  // --- out: [32,64] @ [64,16(9)] + bias, mask, store fp32 ---
  v8f acc0 = {0.f, 0.f, 0.f, 0.f, 0.f, 0.f, 0.f, 0.f};
  v8f acc1 = acc0;
#pragma unroll
  for (int k = 0; k < 2; ++k) {
    v16h a0 = load_a_frag(w.enc, 192, k, lane);
    v16h a1 = load_a_frag(w.enc + 16 * 192, 192, k, lane);
    v16h b  = load_b_frag(ws + OFF_OUT, k, lane);
    acc0 = __builtin_amdgcn_wmma_f32_16x16x32_f16(false, a0, false, b, (short)0, acc0, false, false);
    acc1 = __builtin_amdgcn_wmma_f32_16x16x32_f16(false, a1, false, b, (short)0, acc1, false, false);
  }
  int nl = lane & 15, hi = lane >> 4;
  if (nl < 9) {
    float bv = out_b[nl];
#pragma unroll
    for (int v = 0; v < 8; ++v) {
      int m = v + 8 * hi;
      unsigned mk0 = w.mask[m];
      unsigned mk1 = w.mask[16 + m];
      float x0 = acc0[v] + bv;
      float x1 = acc1[v] + bv;
      x0 = ((mk0 >> nl) & 1u) ? x0 : -__builtin_inff();
      x1 = ((mk1 >> nl) & 1u) ? x1 : -__builtin_inff();
      __builtin_nontemporal_store(x0, &out[(rowBase + (size_t)m) * 9 + nl]);
      __builtin_nontemporal_store(x1, &out[(rowBase + (size_t)(16 + m)) * 9 + nl]);
    }
  }
}

// ---------------------------------------------------------------------------
extern "C" void kernel_launch(void* const* d_in, const int* in_sizes, int n_in,
                              void* d_out, int out_size, void* d_ws, size_t ws_size,
                              hipStream_t stream) {
  const float* bot    = (const float*)d_in[0];
  const float* crew   = (const float*)d_in[1];
  const float* ship   = (const float*)d_in[2];
  const float* bot_w  = (const float*)d_in[3];
  const float* bot_b  = (const float*)d_in[4];
  const float* crew_w = (const float*)d_in[5];
  const float* crew_b = (const float*)d_in[6];
  const float* ship_w = (const float*)d_in[7];
  const float* ship_b = (const float*)d_in[8];
  const float* fc1_w  = (const float*)d_in[9];
  const float* fc1_b  = (const float*)d_in[10];
  const float* fc2_w  = (const float*)d_in[11];
  const float* fc2_b  = (const float*)d_in[12];
  const float* out_w  = (const float*)d_in[13];
  const float* out_b  = (const float*)d_in[14];
  float* out   = (float*)d_out;
  _Float16* ws = (_Float16*)d_ws;

  int B = in_sizes[0] / 121;

  auto pack = [&](const float* wsrc, int off, int K, int N, int Kpad, int Npad) {
    int total  = Kpad * Npad;
    int blocks = (total + 255) / 256;
    pack_weights_kernel<<<dim3(blocks), dim3(256), 0, stream>>>(wsrc, ws + off, K, N, Kpad, Npad);
  };
  pack(bot_w,  OFF_BOT,  121, 64,  128, 64);
  pack(crew_w, OFF_CREW, 121, 64,  128, 64);
  pack(ship_w, OFF_SHIP, 121, 64,  128, 64);
  pack(fc1_w,  OFF_FC1,  192, 128, 192, 128);
  pack(fc2_w,  OFF_FC2,  128, 64,  128, 64);
  pack(out_w,  OFF_OUT,  64,  9,   64,  16);

  int blocks = B / (MROWS * WAVES);
  botnet_kernel<<<dim3(blocks), dim3(32 * WAVES), 0, stream>>>(
      bot, crew, ship, bot_b, crew_b, ship_b, fc1_b, fc2_b, out_b, ws, out);
}
